// RoPEALiBiConformerEncoderLayer_56461640073548
// MI455X (gfx1250) — matrix operand order repaired and checked by reading, not compiled, not run on hardware
//
#include <hip/hip_runtime.h>
#include <hip/hip_bf16.h>
#include <math.h>
#include <stdint.h>

typedef _Float16 h16;
typedef __attribute__((ext_vector_type(16))) _Float16 v16h;
typedef __attribute__((ext_vector_type(8)))  _Float16 v8h;
typedef __attribute__((ext_vector_type(8)))  float    v8f;

#define BB   2
#define LL   2048
#define DD   512
#define HH   16
#define HDD  32
#define FFD  2048
#define KK   5
#define NTOK (BB * LL)

__device__ __forceinline__ v16h cat16(v8h a, v8h b) {
  v16h r;
#pragma unroll
  for (int i = 0; i < 8; ++i) { r[i] = a[i]; r[i + 8] = b[i]; }
  return r;
}

// CDNA5 async global->LDS copy (ASYNCcnt-tracked), 16 bytes per lane.
// vdst = LDS byte offset (low 32 bits of generic shared address), vaddr = global addr.
__device__ __forceinline__ void async_copy_b128(unsigned ldsOff, const void* gptr) {
  asm volatile("global_load_async_to_lds_b128 %0, %1, off"
               :: "v"(ldsOff), "v"((unsigned long long)(uintptr_t)gptr)
               : "memory");
}
__device__ __forceinline__ void wait_async(int pending) {
  if (pending == 0)
    asm volatile("s_wait_asynccnt 0" ::: "memory");
  else
    asm volatile("s_wait_asynccnt 4" ::: "memory");
}

// ---------------------------------------------------------------------------
// f32 -> f16 conversion
// ---------------------------------------------------------------------------
__global__ __launch_bounds__(256) void f32_to_f16_kernel(
    const float* __restrict__ a, h16* __restrict__ o, long n) {
  long i = (long)blockIdx.x * blockDim.x + threadIdx.x;
  if (i < n) o[i] = (h16)a[i];
}

// ---------------------------------------------------------------------------
// Generic WMMA GEMM:  C[M,N] = A[M,K] (f16, row-major) @ W[N,K]^T (f16, row-major)
// Software-pipelined: double-buffered LDS tiles filled with
// global_load_async_to_lds_b128 one K-step ahead of the WMMA compute.
// Epilogue: +biasN[col], +biasM[row], GELU(exact), +resid, write f32 and/or f16.
// Batched via gridDim.z with element strides sA/sW/sC.
// Tiles: 128x128x32, 256 threads (8 waves as 2(M) x 4(N)), each wave 64x32.
// ---------------------------------------------------------------------------
__global__ __launch_bounds__(256) void gemm_wmma_f16(
    const h16* __restrict__ A, const h16* __restrict__ W,
    const float* __restrict__ biasN, const float* __restrict__ biasM,
    const float* __restrict__ resid,
    float* __restrict__ Cf, h16* __restrict__ Ch,
    int M, int N, int K, int act, long sA, long sW, long sC) {
  __shared__ h16 As[2][128 * 40];
  __shared__ h16 Bs[2][128 * 40];

  const long bz = blockIdx.z;
  A += bz * sA;
  W += bz * sW;
  const long cb = bz * sC;

  const int tid   = threadIdx.x;
  const int lane  = tid & 31;
  const int wid   = tid >> 5;
  const int waveM = wid >> 2;   // 0..1 -> 64 rows
  const int waveN = wid & 3;    // 0..3 -> 32 cols
  const int hi    = lane >> 4;
  const int l15   = lane & 15;
  const int blockM = blockIdx.y * 128;
  const int blockN = blockIdx.x * 128;

  const int ldRow = tid >> 1;         // 0..127
  const int ldK   = (tid & 1) * 16;   // 0 or 16

  const h16* gA = A + (long)(blockM + ldRow) * K + ldK;
  const h16* gB = W + (long)(blockN + ldRow) * K + ldK;

  v8f zero8 = {};
  v8f acc[4][2];
#pragma unroll
  for (int i = 0; i < 4; ++i)
#pragma unroll
    for (int j = 0; j < 2; ++j) acc[i][j] = zero8;

  auto issue_tile = [&](int k0, int buf) {
    const unsigned ao = (unsigned)(uintptr_t)&As[buf][ldRow * 40 + ldK];
    const unsigned bo = (unsigned)(uintptr_t)&Bs[buf][ldRow * 40 + ldK];
    async_copy_b128(ao,      gA + k0);
    async_copy_b128(ao + 16, gA + k0 + 8);
    async_copy_b128(bo,      gB + k0);
    async_copy_b128(bo + 16, gB + k0 + 8);
  };

  const int nIter = K >> 5;
  issue_tile(0, 0);

  for (int it = 0; it < nIter; ++it) {
    const int buf = it & 1;
    if (it + 1 < nIter) {
      issue_tile((it + 1) * 32, buf ^ 1);
      wait_async(4);   // first 4 (this tile) complete; next 4 in flight
    } else {
      wait_async(0);
    }
    __syncthreads();   // all waves' current tile resident in LDS

    // A fragment: lane holds row (l15); halves K = [8*hi,8*hi+8) ++ [16+8*hi, +8)
    v16h af[4];
#pragma unroll
    for (int i = 0; i < 4; ++i) {
      const int row = waveM * 64 + i * 16 + l15;
      v8h lo  = *(const v8h*)&As[buf][row * 40 + 8 * hi];
      v8h hi8 = *(const v8h*)&As[buf][row * 40 + 16 + 8 * hi];
      af[i] = cat16(lo, hi8);
    }
    // B fragment: lane holds col (l15); halves K = [8*hi,8*hi+8) ++ [16+8*hi, +8)
    v16h bf[2];
#pragma unroll
    for (int j = 0; j < 2; ++j) {
      const int row = waveN * 32 + j * 16 + l15;
      v8h lo  = *(const v8h*)&Bs[buf][row * 40 + 8 * hi];
      v8h hi8 = *(const v8h*)&Bs[buf][row * 40 + 16 + 8 * hi];
      bf[j] = cat16(lo, hi8);
    }
#pragma unroll
    for (int i = 0; i < 4; ++i)
#pragma unroll
      for (int j = 0; j < 2; ++j)
        acc[i][j] = __builtin_amdgcn_wmma_f32_16x16x32_f16(
            false, af[i], false, bf[j], (short)0, acc[i][j], false, false);
    __syncthreads();   // all waves done reading buf before it is refilled
  }

#pragma unroll
  for (int i = 0; i < 4; ++i)
#pragma unroll
    for (int j = 0; j < 2; ++j)
#pragma unroll
      for (int r = 0; r < 8; ++r) {
        const int row = blockM + waveM * 64 + i * 16 + r + 8 * hi;
        const int col = blockN + waveN * 32 + j * 16 + l15;
        if (row < M && col < N) {
          float v = acc[i][j][r];
          if (biasN) v += biasN[col];
          if (biasM) v += biasM[row];
          if (act == 1) v = 0.5f * v * (1.0f + erff(v * 0.70710678f));
          const long o = cb + (long)row * N + col;
          if (resid) v += resid[o];
          if (Cf) Cf[o] = v;
          if (Ch) Ch[o] = (h16)v;
        }
      }
}

// ---------------------------------------------------------------------------
// RoPE + head packing. Mirrors the reference's even/odd slicing of the
// concatenated [cos|sin] table. q,k -> [B,H,L,HD] f16; v -> transposed
// [B,H,HD,L] f16 so attention B-fragments are contiguous.
// ---------------------------------------------------------------------------
__global__ __launch_bounds__(256) void rope_pack_kernel(
    const h16* __restrict__ qf, const h16* __restrict__ kf,
    const h16* __restrict__ vf, h16* __restrict__ qro,
    h16* __restrict__ kro, h16* __restrict__ vt) {
  long idx = (long)blockIdx.x * blockDim.x + threadIdx.x;
  if (idx >= (long)BB * HH * LL) return;
  const int l = (int)(idx % LL);
  const int h = (int)((idx / LL) % HH);
  const int b = (int)(idx / ((long)LL * HH));
  const h16* qp = qf + ((long)(b * LL + l)) * DD + h * HDD;
  const h16* kp = kf + ((long)(b * LL + l)) * DD + h * HDD;
  const h16* vp = vf + ((long)(b * LL + l)) * DD + h * HDD;
  h16* qo = qro + ((long)(b * HH + h) * LL + l) * HDD;
  h16* ko = kro + ((long)(b * HH + h) * LL + l) * HDD;
  const float fl = (float)l;
#pragma unroll
  for (int j = 0; j < 16; ++j) {
    float cj, sj;
    if (j < 8) {
      cj = __cosf(fl * __powf(10000.f, -(float)(2 * j) / 16.f));
      sj = __cosf(fl * __powf(10000.f, -(float)(2 * j + 1) / 16.f));
    } else {
      const int jj = j - 8;
      cj = __sinf(fl * __powf(10000.f, -(float)(2 * jj) / 16.f));
      sj = __sinf(fl * __powf(10000.f, -(float)(2 * jj + 1) / 16.f));
    }
    const float qe = (float)qp[2 * j], qo_ = (float)qp[2 * j + 1];
    const float ke = (float)kp[2 * j], ko_ = (float)kp[2 * j + 1];
    qo[j]      = (h16)(qe * cj - qo_ * sj);
    qo[16 + j] = (h16)(qe * sj + qo_ * cj);
    ko[j]      = (h16)(ke * cj - ko_ * sj);
    ko[16 + j] = (h16)(ke * sj + ko_ * cj);
  }
  h16* vo = vt + ((long)(b * HH + h) * HDD) * LL + l;
#pragma unroll
  for (int d = 0; d < HDD; ++d) vo[(long)d * LL] = vp[d];
}

// ---------------------------------------------------------------------------
// Flash attention with ALiBi. Grid (L/128, H, B), 256 threads.
// Each wave owns 16 query rows; online softmax over 32-key blocks.
// S tiles and PV accumulation both via v_wmma_f32_16x16x32_f16.
// ---------------------------------------------------------------------------
__global__ __launch_bounds__(256) void flash_attn_wmma(
    const h16* __restrict__ Q, const h16* __restrict__ Km,
    const h16* __restrict__ Vt, h16* __restrict__ Ctx) {
  __shared__ h16 plds[8][16 * 40];  // per-wave P transpose tile (padded)
  const int tid  = threadIdx.x;
  const int lane = tid & 31;
  const int wid  = tid >> 5;
  const int hi   = lane >> 4;
  const int l15  = lane & 15;
  const int b = blockIdx.z, h = blockIdx.y;
  const int qBase = blockIdx.x * 128 + wid * 16;

  const h16* q  = Q  + (long)(b * HH + h) * LL * HDD;
  const h16* k  = Km + (long)(b * HH + h) * LL * HDD;
  const h16* vt = Vt + (long)(b * HH + h) * HDD * LL;

  const float scale = 0.17677669529663687f;  // 1/sqrt(32)
  const float slope = exp2f(-exp2f(-(log2f((float)HH) - 3.0f)) * (float)h);

  v16h aq;
  {
    const h16* qr = q + (long)(qBase + l15) * HDD;
    v8h q0 = *(const v8h*)(qr + 8 * hi);
    v8h q1 = *(const v8h*)(qr + 16 + 8 * hi);
    aq = cat16(q0, q1);
  }

  float mrow[8], lrow[8];
#pragma unroll
  for (int r = 0; r < 8; ++r) { mrow[r] = -3.0e38f; lrow[r] = 0.f; }
  v8f zero8 = {};
  v8f vacc[2];
  vacc[0] = zero8; vacc[1] = zero8;

  h16* P = &plds[wid][0];

  for (int kb = 0; kb < LL; kb += 32) {
    v8f s0t, s1t;
    {
      const h16* kr = k + (long)(kb + l15) * HDD + 8 * hi;
      v16h bk = cat16(*(const v8h*)kr, *(const v8h*)(kr + 16));
      s0t = __builtin_amdgcn_wmma_f32_16x16x32_f16(false, aq, false, bk,
                                                   (short)0, zero8, false, false);
    }
    {
      const h16* kr = k + (long)(kb + 16 + l15) * HDD + 8 * hi;
      v16h bk = cat16(*(const v8h*)kr, *(const v8h*)(kr + 16));
      s1t = __builtin_amdgcn_wmma_f32_16x16x32_f16(false, aq, false, bk,
                                                   (short)0, zero8, false, false);
    }
    float p0[8], p1[8], alpha[8];
#pragma unroll
    for (int r = 0; r < 8; ++r) {
      const float rowq = (float)(qBase + r + 8 * hi);
      float s0 = s0t[r] * scale + slope * fmaxf(rowq - (float)(kb + l15), 0.f);
      float s1 = s1t[r] * scale + slope * fmaxf(rowq - (float)(kb + 16 + l15), 0.f);
      float tm = fmaxf(s0, s1);
      tm = fmaxf(tm, __shfl_xor(tm, 1));
      tm = fmaxf(tm, __shfl_xor(tm, 2));
      tm = fmaxf(tm, __shfl_xor(tm, 4));
      tm = fmaxf(tm, __shfl_xor(tm, 8));
      const float mnew = fmaxf(mrow[r], tm);
      const float a = __expf(mrow[r] - mnew);
      p0[r] = __expf(s0 - mnew);
      p1[r] = __expf(s1 - mnew);
      float ts = p0[r] + p1[r];
      ts += __shfl_xor(ts, 1);
      ts += __shfl_xor(ts, 2);
      ts += __shfl_xor(ts, 4);
      ts += __shfl_xor(ts, 8);
      lrow[r] = lrow[r] * a + ts;
      mrow[r] = mnew;
      alpha[r] = a;
    }
#pragma unroll
    for (int j = 0; j < 2; ++j)
#pragma unroll
      for (int r = 0; r < 8; ++r) vacc[j][r] *= alpha[r];

    // C-layout P -> LDS -> A-fragment layout (wave-private; DS is in-order)
#pragma unroll
    for (int r = 0; r < 8; ++r) {
      P[(r + 8 * hi) * 40 + l15]      = (h16)p0[r];
      P[(r + 8 * hi) * 40 + 16 + l15] = (h16)p1[r];
    }
    v16h pa;
    {
      v8h a0 = *(const v8h*)&P[l15 * 40 + 8 * hi];
      v8h a1 = *(const v8h*)&P[l15 * 40 + 16 + 8 * hi];
      pa = cat16(a0, a1);
    }
#pragma unroll
    for (int j = 0; j < 2; ++j) {
      const h16* vr = vt + (long)(j * 16 + l15) * LL + kb + 8 * hi;
      v16h bv = cat16(*(const v8h*)vr, *(const v8h*)(vr + 16));
      vacc[j] = __builtin_amdgcn_wmma_f32_16x16x32_f16(false, pa, false, bv,
                                                       (short)0, vacc[j], false, false);
    }
  }

#pragma unroll
  for (int j = 0; j < 2; ++j)
#pragma unroll
    for (int r = 0; r < 8; ++r) {
      const int l = qBase + r + 8 * hi;
      const int d = h * HDD + j * 16 + l15;
      Ctx[((long)(b * LL + l)) * DD + d] = (h16)(vacc[j][r] / lrow[r]);
    }
}

// ---------------------------------------------------------------------------
// Fused LayerNorm: t = in (+ addIn); x = LN(t)*g + b; out = (preRes +) x.
// One block per row of D=512; optional f32 / f16 / transposed-f16 outputs.
// ---------------------------------------------------------------------------
__global__ __launch_bounds__(256) void ln_fused_kernel(
    const float* __restrict__ in, const float* __restrict__ addIn,
    const float* __restrict__ preRes, const float* __restrict__ g,
    const float* __restrict__ bb, float* __restrict__ outF,
    h16* __restrict__ outH, h16* __restrict__ outT) {
  __shared__ float sred[256];
  const int row = blockIdx.x;
  const int tid = threadIdx.x;
  const long base = (long)row * DD;
  float v0 = in[base + tid];
  float v1 = in[base + tid + 256];
  if (addIn) { v0 += addIn[base + tid]; v1 += addIn[base + tid + 256]; }
  sred[tid] = v0 + v1;
  __syncthreads();
  for (int off = 128; off > 0; off >>= 1) {
    if (tid < off) sred[tid] += sred[tid + off];
    __syncthreads();
  }
  const float mean = sred[0] * (1.0f / DD);
  __syncthreads();
  const float d0 = v0 - mean, d1 = v1 - mean;
  sred[tid] = d0 * d0 + d1 * d1;
  __syncthreads();
  for (int off = 128; off > 0; off >>= 1) {
    if (tid < off) sred[tid] += sred[tid + off];
    __syncthreads();
  }
  const float inv = rsqrtf(sred[0] * (1.0f / DD) + 1e-5f);
  float o0 = d0 * inv * g[tid] + bb[tid];
  float o1 = d1 * inv * g[tid + 256] + bb[tid + 256];
  if (preRes) { o0 += preRes[base + tid]; o1 += preRes[base + tid + 256]; }
  if (outF) { outF[base + tid] = o0; outF[base + tid + 256] = o1; }
  if (outH) { outH[base + tid] = (h16)o0; outH[base + tid + 256] = (h16)o1; }
  if (outT) {
    const int bi = row / LL, l = row % LL;
    outT[((long)bi * DD + tid) * LL + l] = (h16)o0;
    outT[((long)bi * DD + tid + 256) * LL + l] = (h16)o1;
  }
}

// ---------------------------------------------------------------------------
// GLU: out = a * sigmoid(g), halves of the pw1 output
// ---------------------------------------------------------------------------
__global__ __launch_bounds__(256) void glu_kernel(const float* __restrict__ in,
                                                  float* __restrict__ out) {
  long idx = (long)blockIdx.x * blockDim.x + threadIdx.x;
  if (idx >= (long)NTOK * DD) return;
  const long row = idx / DD;
  const int c = (int)(idx % DD);
  const float a  = in[row * (2 * DD) + c];
  const float gt = in[row * (2 * DD) + DD + c];
  out[idx] = a * (1.0f / (1.0f + __expf(-gt)));
}

// ---------------------------------------------------------------------------
// Depthwise conv (K=5, pad 2) + folded BN + hardswish -> f16
// ---------------------------------------------------------------------------
__global__ __launch_bounds__(256) void dwconv_kernel(
    const float* __restrict__ x, const float* __restrict__ wgt,
    const float* __restrict__ bias, const float* __restrict__ bn_g,
    const float* __restrict__ bn_b, h16* __restrict__ out) {
  long idx = (long)blockIdx.x * blockDim.x + threadIdx.x;
  if (idx >= (long)NTOK * DD) return;
  const int d = (int)(idx % DD);
  const long bl = idx / DD;
  const int l = (int)(bl % LL);
  const long b = bl / LL;
  float acc = bias[d];
#pragma unroll
  for (int t = 0; t < KK; ++t) {
    const int ll = l + t - 2;
    if (ll >= 0 && ll < LL)
      acc += x[((long)b * LL + ll) * DD + d] * wgt[d * KK + t];
  }
  acc = acc * (bn_g[d] * rsqrtf(1.0f + 1e-5f)) + bn_b[d];
  acc = acc * fminf(fmaxf(acc + 3.0f, 0.0f), 6.0f) * (1.0f / 6.0f);
  out[idx] = (h16)acc;
}

// ---------------------------------------------------------------------------
extern "C" void kernel_launch(void* const* d_in, const int* in_sizes, int n_in,
                              void* d_out, int out_size, void* d_ws, size_t ws_size,
                              hipStream_t stream) {
  (void)in_sizes; (void)n_in; (void)out_size; (void)ws_size;
  const float* src     = (const float*)d_in[0];
  const float* Wq      = (const float*)d_in[1];
  const float* bq      = (const float*)d_in[2];
  const float* Wk      = (const float*)d_in[3];
  const float* bk      = (const float*)d_in[4];
  const float* Wv      = (const float*)d_in[5];
  const float* bv      = (const float*)d_in[6];
  const float* Wo      = (const float*)d_in[7];
  const float* bo      = (const float*)d_in[8];
  const float* ln_g    = (const float*)d_in[9];
  const float* ln_b    = (const float*)d_in[10];
  const float* pw1_w   = (const float*)d_in[11];
  const float* pw1_b   = (const float*)d_in[12];
  const float* dw_w    = (const float*)d_in[13];
  const float* dw_b    = (const float*)d_in[14];
  const float* bn_g    = (const float*)d_in[15];
  const float* bn_b    = (const float*)d_in[16];
  const float* pw2_w   = (const float*)d_in[17];
  const float* pw2_b   = (const float*)d_in[18];
  const float* proj_w  = (const float*)d_in[19];
  const float* proj_b  = (const float*)d_in[20];
  const float* proj2_w = (const float*)d_in[21];
  const float* proj2_b = (const float*)d_in[22];
  const float* W1      = (const float*)d_in[23];
  const float* b1      = (const float*)d_in[24];
  const float* W2      = (const float*)d_in[25];
  const float* b2      = (const float*)d_in[26];
  const float* n1_g    = (const float*)d_in[27];
  const float* n1_b    = (const float*)d_in[28];
  const float* n2_g    = (const float*)d_in[29];
  const float* n2_b    = (const float*)d_in[30];

  char* wp = (char*)d_ws;
  auto alloc = [&](size_t bytes) -> void* {
    void* p = (void*)wp;
    wp += (bytes + 255) & ~(size_t)255;
    return p;
  };

  const long ND = (long)NTOK * DD;  // 2M elements

  // f16 copies of GEMM operands
  h16* srcH   = (h16*)alloc(ND * 2);
  h16* WqH    = (h16*)alloc((long)DD * DD * 2);
  h16* WkH    = (h16*)alloc((long)DD * DD * 2);
  h16* WvH    = (h16*)alloc((long)DD * DD * 2);
  h16* WoH    = (h16*)alloc((long)DD * DD * 2);
  h16* pw1H   = (h16*)alloc((long)2 * DD * DD * 2);
  h16* pw2H   = (h16*)alloc((long)DD * DD * 2);
  h16* projH  = (h16*)alloc((long)LL * LL * 2);
  h16* proj2H = (h16*)alloc((long)DD * DD * 2);
  h16* W1H    = (h16*)alloc((long)FFD * DD * 2);
  h16* W2H    = (h16*)alloc((long)DD * FFD * 2);
  // activations
  h16*  qH    = (h16*)alloc(ND * 2);
  h16*  kH    = (h16*)alloc(ND * 2);
  h16*  vH    = (h16*)alloc(ND * 2);
  h16*  qro   = (h16*)alloc(ND * 2);
  h16*  kro   = (h16*)alloc(ND * 2);
  h16*  vtH   = (h16*)alloc(ND * 2);
  h16*  ctxH  = (h16*)alloc(ND * 2);
  float* attnF = (float*)alloc(ND * 4);
  float* x0F   = (float*)alloc(ND * 4);
  h16*  x0tH  = (h16*)alloc(ND * 2);
  h16*  xlnH  = (h16*)alloc(ND * 2);
  float* pw1F  = (float*)alloc((long)NTOK * 2 * DD * 4);
  float* gluF  = (float*)alloc(ND * 4);
  h16*  convH = (h16*)alloc(ND * 2);
  float* convF = (float*)alloc(ND * 4);
  h16*  pH    = (h16*)alloc(ND * 2);
  float* yF    = (float*)alloc(ND * 4);
  h16*  yH    = (h16*)alloc(ND * 2);
  h16*  hH    = (h16*)alloc((long)NTOK * FFD * 2);
  float* zF    = (float*)alloc(ND * 4);

  auto cvt = [&](const float* a, h16* o, long n) {
    f32_to_f16_kernel<<<dim3((unsigned)((n + 255) / 256)), 256, 0, stream>>>(a, o, n);
  };
  cvt(src, srcH, ND);
  cvt(Wq, WqH, (long)DD * DD);
  cvt(Wk, WkH, (long)DD * DD);
  cvt(Wv, WvH, (long)DD * DD);
  cvt(Wo, WoH, (long)DD * DD);
  cvt(pw1_w, pw1H, (long)2 * DD * DD);
  cvt(pw2_w, pw2H, (long)DD * DD);
  cvt(proj_w, projH, (long)LL * LL);
  cvt(proj2_w, proj2H, (long)DD * DD);
  cvt(W1, W1H, (long)FFD * DD);
  cvt(W2, W2H, (long)DD * FFD);

  const dim3 blk(256);
  // QKV projections (f16 outputs only)
  gemm_wmma_f16<<<dim3(DD / 128, NTOK / 128, 1), blk, 0, stream>>>(
      srcH, WqH, bq, nullptr, nullptr, nullptr, qH, NTOK, DD, DD, 0, 0, 0, 0);
  gemm_wmma_f16<<<dim3(DD / 128, NTOK / 128, 1), blk, 0, stream>>>(
      srcH, WkH, bk, nullptr, nullptr, nullptr, kH, NTOK, DD, DD, 0, 0, 0, 0);
  gemm_wmma_f16<<<dim3(DD / 128, NTOK / 128, 1), blk, 0, stream>>>(
      srcH, WvH, bv, nullptr, nullptr, nullptr, vH, NTOK, DD, DD, 0, 0, 0, 0);

  rope_pack_kernel<<<dim3((BB * HH * LL + 255) / 256), blk, 0, stream>>>(
      qH, kH, vH, qro, kro, vtH);

  flash_attn_wmma<<<dim3(LL / 128, HH, BB), blk, 0, stream>>>(qro, kro, vtH, ctxH);

  // attn_out = ctx @ Wo^T + bo
  gemm_wmma_f16<<<dim3(DD / 128, NTOK / 128, 1), blk, 0, stream>>>(
      ctxH, WoH, bo, nullptr, nullptr, attnF, nullptr, NTOK, DD, DD, 0, 0, 0, 0);

  // x0 = src + LN(attn_out; n1)  (also transposed f16 for the proj_w GEMM)
  ln_fused_kernel<<<dim3(NTOK), blk, 0, stream>>>(
      attnF, nullptr, src, n1_g, n1_b, x0F, nullptr, x0tH);
  // xln = LN(x0; ln)
  ln_fused_kernel<<<dim3(NTOK), blk, 0, stream>>>(
      x0F, nullptr, nullptr, ln_g, ln_b, nullptr, xlnH, nullptr);

  // pw1 -> GLU -> depthwise conv + BN + hardswish -> pw2 (+x0 residual)
  gemm_wmma_f16<<<dim3((2 * DD) / 128, NTOK / 128, 1), blk, 0, stream>>>(
      xlnH, pw1H, pw1_b, nullptr, nullptr, pw1F, nullptr, NTOK, 2 * DD, DD, 0, 0, 0, 0);
  glu_kernel<<<dim3((unsigned)((ND + 255) / 256)), blk, 0, stream>>>(pw1F, gluF);
  dwconv_kernel<<<dim3((unsigned)((ND + 255) / 256)), blk, 0, stream>>>(
      gluF, dw_w, dw_b, bn_g, bn_b, convH);
  gemm_wmma_f16<<<dim3(DD / 128, NTOK / 128, 1), blk, 0, stream>>>(
      convH, pw2H, pw2_b, nullptr, x0F, convF, nullptr, NTOK, DD, DD, 0, 0, 0, 0);

  // p[b] = proj_w @ x0[b] (contraction over sequence) + proj_b[m]  (batched)
  gemm_wmma_f16<<<dim3(DD / 128, LL / 128, BB), blk, 0, stream>>>(
      projH, x0tH, nullptr, proj_b, nullptr, nullptr, pH, LL, DD, LL, 0,
      0, (long)DD * LL, (long)LL * DD);
  // y = p @ proj2_w^T + proj2_b + conv
  gemm_wmma_f16<<<dim3(DD / 128, NTOK / 128, 1), blk, 0, stream>>>(
      pH, proj2H, proj2_b, nullptr, convF, yF, yH, NTOK, DD, DD, 0, 0, 0, 0);

  // FFN: h = GELU(y @ W1^T + b1); z = y + h @ W2^T + b2
  gemm_wmma_f16<<<dim3(FFD / 128, NTOK / 128, 1), blk, 0, stream>>>(
      yH, W1H, b1, nullptr, nullptr, nullptr, hH, NTOK, FFD, DD, 1, 0, 0, 0);
  gemm_wmma_f16<<<dim3(DD / 128, NTOK / 128, 1), blk, 0, stream>>>(
      hH, W2H, b2, nullptr, yF, zF, nullptr, NTOK, DD, FFD, 0, 0, 0, 0);

  // out = LN(z; n2)
  ln_fused_kernel<<<dim3(NTOK), blk, 0, stream>>>(
      zF, nullptr, nullptr, n2_g, n2_b, (float*)d_out, nullptr, nullptr);
}
